// TSIrregularEncoder_80092550135876
// MI455X (gfx1250) — compile-verified, matrix operand order checked by reading, not run on hardware
//
#include <hip/hip_runtime.h>
#include <hip/hip_bf16.h>
#include <math.h>

// ---------------------------------------------------------------------------
// mTAND irregular-time-series encoder, fused for MI455X (gfx1250, wave32).
// All GEMMs via v_wmma_f32_16x16x32_f16 (f32 accumulate); attention operands
// bulk-staged into LDS with global_load_async_to_lds_b128 (ASYNCcnt path).
// Workspace use: ~4.4 MB of d_ws (L2-resident on a 192MB-L2 part).
// ---------------------------------------------------------------------------

static constexpr int NB = 8, NL = 256, ND = 32, NE = 128, NH = 8, NQ = 128;
static constexpr int NHID = 256, NV = 64, NEK = 16;
static constexpr int KP = 32;  // K-padded row length for q/k tiles

typedef __attribute__((ext_vector_type(16))) _Float16 v16h;
typedef __attribute__((ext_vector_type(8)))  float    v8f;

// (lane-half j, lane-group hi) -> K index for 16-bit 16x32 A / 32x16 B tiles
// (CDNA5 ISA 7.12.2: lanes 0-15 hold K {0-7,16-23}, lanes 16-31 hold K {8-15,24-31})
__device__ __forceinline__ int kmap(int j, int hi) {
  return (j < 8) ? (hi * 8 + j) : (16 + hi * 8 + (j - 8));
}

// A tile: 16x32 (MxK), row-major f16 source, leading dim ldm
__device__ __forceinline__ v16h load_a_h(const _Float16* p, int ldm) {
  int lane = threadIdx.x & 31, r = lane & 15, hi = lane >> 4;
  v16h a;
#pragma unroll
  for (int j = 0; j < 16; ++j) a[j] = p[r * ldm + kmap(j, hi)];
  return a;
}

// A tile from f32 LDS (convert to f16 on the fly)
__device__ __forceinline__ v16h load_a_f32(const float* p, int ldm) {
  int lane = threadIdx.x & 31, r = lane & 15, hi = lane >> 4;
  v16h a;
#pragma unroll
  for (int j = 0; j < 16; ++j) a[j] = (_Float16)p[r * ldm + kmap(j, hi)];
  return a;
}

// B tile: 32x16 (KxN); element (k,c) at p[k*sk + c*sc]
__device__ __forceinline__ v16h load_b_gen(const _Float16* p, int sk, int sc) {
  int lane = threadIdx.x & 31, c = lane & 15, hi = lane >> 4;
  v16h bm;
#pragma unroll
  for (int j = 0; j < 16; ++j) bm[j] = p[kmap(j, hi) * sk + c * sc];
  return bm;
}

__device__ __forceinline__ v8f wmma16(v16h a, v16h b, v8f c) {
  return __builtin_amdgcn_wmma_f32_16x16x32_f16(false, a, false, b,
                                                (short)0, c, false, false);
}

// Store f32 16x16 D tile (VGPR r: lanes0-15 M=r, lanes16-31 M=r+8; N=lane&15)
__device__ __forceinline__ void store_d_f32(float* p, int ld, v8f d) {
  int lane = threadIdx.x & 31, n = lane & 15, hi = lane >> 4;
#pragma unroll
  for (int r = 0; r < 8; ++r) p[(r + 8 * hi) * ld + n] = d[r];
}

// gfx1250 async bulk copy: 16B global -> LDS, tracked by ASYNCcnt
__device__ __forceinline__ void async_copy_b128(unsigned lds_addr, const void* g) {
  asm volatile("global_load_async_to_lds_b128 %0, %1, off"
               :: "v"(lds_addr), "v"(g) : "memory");
}
__device__ __forceinline__ void wait_async0() {
  asm volatile("s_wait_asynccnt 0x0" ::: "memory");
}

// ---------------------------------------------------------------------------
// K0: per-batch time normalization, masked-value matrix, colmeans, qt output
// ---------------------------------------------------------------------------
__global__ __launch_bounds__(256) void k_prep(
    const float* __restrict__ values, const int* __restrict__ mask,
    const float* __restrict__ times, float* __restrict__ tn,
    float* __restrict__ hasf, float* __restrict__ colmean,
    _Float16* __restrict__ mcat, float* __restrict__ qt_out) {
  int b = blockIdx.x, t = threadIdx.x;  // t == l
  __shared__ float smin[256], smax[256];
  __shared__ int sobs[256];

  const float* vrow = values + (size_t)(b * NL + t) * ND;
  const int*   mrow = mask   + (size_t)(b * NL + t) * ND;
  int any = 0;
#pragma unroll
  for (int d = 0; d < ND; ++d) {
    int mi = mrow[d];
    any |= mi;
    float mf = (float)mi;
    mcat[(size_t)(b * NL + t) * NV + d]      = (_Float16)(mf * vrow[d]);
    mcat[(size_t)(b * NL + t) * NV + ND + d] = (_Float16)mf;
  }
  float tv = times[b * NL + t];
  smin[t] = any ? tv : INFINITY;
  smax[t] = any ? tv : -INFINITY;
  sobs[t] = any;
  __syncthreads();
  for (int s = 128; s > 0; s >>= 1) {
    if (t < s) {
      smin[t] = fminf(smin[t], smin[t + s]);
      smax[t] = fmaxf(smax[t], smax[t + s]);
      sobs[t] |= sobs[t + s];
    }
    __syncthreads();
  }
  int   has  = sobs[0];
  float tmin = has ? smin[0] : 0.0f;
  float tmax = has ? smax[0] : 1.0f;
  float tnv  = (tv - tmin) / fmaxf(tmax - tmin, 1e-6f);
  tn[b * NL + t] = any ? tnv : 0.0f;
  if (t == 0) hasf[b] = has ? 1.0f : 0.0f;

  // uniform-attention fallback: mean over L of vinp = [values | maskf]
  if (t < NV) {
    int d = t & (ND - 1);
    bool ismask = t >= ND;
    float s = 0.0f;
    for (int l = 0; l < NL; ++l) {
      s += ismask ? (float)mask[(size_t)(b * NL + l) * ND + d]
                  : values[(size_t)(b * NL + l) * ND + d];
    }
    colmean[b * NV + t] = s / (float)NL;
  }
  if (t < NQ) qt_out[b * NQ + t] = (float)t / (float)(NQ - 1);
}

// ---------------------------------------------------------------------------
// K1: convert weights to f16 once per launch
// ---------------------------------------------------------------------------
__global__ __launch_bounds__(256) void k_cvt(
    const float* __restrict__ wq, const float* __restrict__ wk,
    const float* __restrict__ wo, _Float16* __restrict__ wqh,
    _Float16* __restrict__ wkh, _Float16* __restrict__ woh) {
  int i = blockIdx.x * 256 + threadIdx.x;
  if (i < NE * NE) { wqh[i] = (_Float16)wq[i]; wkh[i] = (_Float16)wk[i]; }
  if (i < NHID * NH * NV) woh[i] = (_Float16)wo[i];
}

// ---------------------------------------------------------------------------
// K2: time embedding rows (B*L key rows then Q query rows), f16 output
// ---------------------------------------------------------------------------
__global__ __launch_bounds__(128) void k_embed(
    const float* __restrict__ tn, const float* __restrict__ w_per,
    const float* __restrict__ b_per, const float* __restrict__ w_lin,
    const float* __restrict__ b_lin, _Float16* __restrict__ tkh,
    _Float16* __restrict__ tqh) {
  int r = blockIdx.x, e = threadIdx.x;
  float t;
  _Float16* dst;
  if (r < NB * NL) { t = tn[r]; dst = tkh + (size_t)r * NE; }
  else { int rq = r - NB * NL; t = (float)rq / (float)(NQ - 1); dst = tqh + (size_t)rq * NE; }
  float v = (e == 0) ? (t * w_lin[0] + b_lin[0])
                     : sinf(t * w_per[e - 1] + b_per[e - 1]);
  dst[e] = (_Float16)v;
}

// ---------------------------------------------------------------------------
// K3: projection GEMM  out = A[M,128] @ W^T + bias  (one wave per 16x16 tile)
// Output stored K-padded [row][KP=32] (upper 16 zero) so attention can
// bulk-stage it with async B128 copies, no pad pass needed.
// MODE 0: K-proj -> kh[B,H,L,KP]; MODE 1: Q-proj -> qh[H,Q,KP] (pre-scaled)
// ---------------------------------------------------------------------------
template <int MODE>
__global__ __launch_bounds__(256) void k_proj(
    const _Float16* __restrict__ A, const _Float16* __restrict__ W,
    const float* __restrict__ bias, _Float16* __restrict__ outp,
    int M, float scale) {
  int tile = blockIdx.x * 8 + (threadIdx.x >> 5);
  int ntiles = (M / 16) * (NE / 16);
  if (tile >= ntiles) return;
  int tm = tile / (NE / 16), tnn = tile % (NE / 16);
  int m0 = tm * 16, n0 = tnn * 16;
  v8f acc = {};
#pragma unroll
  for (int kk = 0; kk < NE; kk += 32) {
    v16h a  = load_a_h(A + (size_t)m0 * NE + kk, NE);
    v16h bm = load_b_gen(W + (size_t)n0 * NE + kk, 1, NE);  // B[k][c]=W[n0+c][kk+k]
    acc = wmma16(a, bm, acc);
  }
  int lane = threadIdx.x & 31, n = n0 + (lane & 15), hi = lane >> 4;
  int h = n >> 4, ek = n & (NEK - 1);
#pragma unroll
  for (int r = 0; r < 8; ++r) {
    int m = m0 + r + 8 * hi;
    float v = (acc[r] + bias[n]) * scale;
    size_t row;
    if (MODE == 0) {  // m = b*L + l
      int bb = m >> 8, l = m & (NL - 1);
      row = (size_t)((bb * NH + h) * NL + l);
    } else {          // m = q
      row = (size_t)(h * NQ + m);
    }
    outp[row * KP + ek]       = (_Float16)v;
    outp[row * KP + NEK + ek] = (_Float16)0.0f;  // zero pad (K 16..31)
  }
}

// ---------------------------------------------------------------------------
// K4: fused attention per (b,h): async-stage q/k/mcat to LDS ->
//     scores WMMA -> rowmax/exp -> ctx WMMA -> divide fix-up
// LDS: 8K(q) + 16K(k) + 128K(scores f32) + 1K(rowmax) + 32K(mcat/gemm-out)
// ---------------------------------------------------------------------------
union CtxShm {
  _Float16 mcatl[NL * NV];  // staged [L,64] value/mask matrix (32KB)
  float    go[NQ * NV];     // raw W@Mcat output (32KB), used after mcatl dies
};

__global__ __launch_bounds__(256) void k_attn(
    const _Float16* __restrict__ qh, const _Float16* __restrict__ kh,
    const _Float16* __restrict__ mcat, const float* __restrict__ colmean,
    _Float16* __restrict__ ctxh) {
  int bh = blockIdx.x, b = bh / NH, h = bh % NH;
  int t = threadIdx.x, wave = t >> 5;
  __shared__ _Float16 qs[NQ * KP];   // 8KB
  __shared__ _Float16 ks[NL * KP];   // 16KB
  __shared__ float    sc[NQ * NL];   // 128KB scores, then exp-weights
  __shared__ float    rmax_s[256];   // 1KB
  __shared__ CtxShm   cs;            // 32KB

  // ---- stage operands into LDS with gfx1250 async B128 copies ----
  const char* qsrc = (const char*)(qh + (size_t)h * NQ * KP);
  const char* ksrc = (const char*)(kh + (size_t)(b * NH + h) * NL * KP);
  const char* msrc = (const char*)(mcat + (size_t)b * NL * NV);
  unsigned qdst = (unsigned)(size_t)qs;
  unsigned kdst = (unsigned)(size_t)ks;
  unsigned mdst = (unsigned)(size_t)cs.mcatl;
#pragma unroll
  for (int ofs = 0; ofs < NQ * KP * 2; ofs += 256 * 16)
    async_copy_b128(qdst + ofs + t * 16, qsrc + ofs + t * 16);
#pragma unroll
  for (int ofs = 0; ofs < NL * KP * 2; ofs += 256 * 16)
    async_copy_b128(kdst + ofs + t * 16, ksrc + ofs + t * 16);
#pragma unroll
  for (int ofs = 0; ofs < NL * NV * 2; ofs += 256 * 16)
    async_copy_b128(mdst + ofs + t * 16, msrc + ofs + t * 16);
  wait_async0();
  __syncthreads();

  // ---- scores[Q,L] = q @ k^T (q pre-scaled by 1/sqrt(EK)); 128 tiles ----
  for (int tile = wave; tile < (NQ / 16) * (NL / 16); tile += 8) {
    int tm = tile / (NL / 16), tl = tile % (NL / 16);
    int m0 = tm * 16, n0 = tl * 16;
    v16h a  = load_a_h(qs + m0 * KP, KP);
    v16h bm = load_b_gen(ks + n0 * KP, 1, KP);  // B[k][c] = k[n0+c][k]
    v8f acc = {};
    acc = wmma16(a, bm, acc);
    store_d_f32(&sc[m0 * NL + n0], NL, acc);
  }
  __syncthreads();

  // ---- row max + exp in place (2 threads per row) ----
  {
    int q = t >> 1, half = t & 1;
    float* row = &sc[q * NL + half * (NL / 2)];
    float m = -INFINITY;
    for (int l = 0; l < NL / 2; ++l) m = fmaxf(m, row[l]);
    rmax_s[t] = m;
    __syncthreads();
    m = fmaxf(rmax_s[q * 2], rmax_s[q * 2 + 1]);
    for (int l = 0; l < NL / 2; ++l) row[l] = expf(row[l] - m);
  }
  __syncthreads();

  // ---- ctx_raw[Q,64] = W[Q,L] @ Mcat[L,64]; 32 tiles -> 4/wave, K=256 ----
  v8f accs[4];
#pragma unroll
  for (int i = 0; i < 4; ++i) {
    int tile = wave * 4 + i;
    int m0 = (tile >> 2) * 16, v0 = (tile & 3) * 16;
    v8f acc = {};
#pragma unroll
    for (int kk = 0; kk < NL; kk += 32) {
      v16h a  = load_a_f32(&sc[m0 * NL + kk], NL);
      v16h bm = load_b_gen(cs.mcatl + kk * NV + v0, NV, 1);
      acc = wmma16(a, bm, acc);
    }
    accs[i] = acc;
  }
  __syncthreads();  // all mcatl reads done before union is overwritten
#pragma unroll
  for (int i = 0; i < 4; ++i) {
    int tile = wave * 4 + i;
    int m0 = (tile >> 2) * 16, v0 = (tile & 3) * 16;
    store_d_f32(&cs.go[m0 * NV + v0], NV, accs[i]);
  }
  __syncthreads();

  // ---- ctx = num/den (den = W@mask cols 32..63), fallback to colmean ----
  for (int i = t; i < NQ * NV; i += 256) {
    int q = i >> 6, v = i & (NV - 1);
    float den = cs.go[q * NV + ND + (v & (ND - 1))];
    float num = cs.go[i];
    float val = (den > 1e-20f) ? (num / den) : colmean[b * NV + v];
    ctxh[(size_t)(b * NQ + q) * (NH * NV) + h * NV + v] = (_Float16)val;
  }
}

// ---------------------------------------------------------------------------
// K5: output GEMM  seqpre[B,Q,HID] = ctx[B,Q,512] @ wo^T + bo
// ---------------------------------------------------------------------------
__global__ __launch_bounds__(256) void k_ogemm(
    const _Float16* __restrict__ ctxh, const _Float16* __restrict__ woh,
    const float* __restrict__ bo, float* __restrict__ seqpre) {
  int tile = blockIdx.x * 8 + (threadIdx.x >> 5);
  constexpr int KDIM = NH * NV;                   // 512
  constexpr int per_b = (NQ / 16) * (NHID / 16);  // 128
  int b = tile / per_b, r = tile % per_b;
  int m0 = (r / (NHID / 16)) * 16, n0 = (r % (NHID / 16)) * 16;
  const _Float16* A = ctxh + (size_t)b * NQ * KDIM;
  v8f acc = {};
#pragma unroll
  for (int kk = 0; kk < KDIM; kk += 32) {
    v16h a  = load_a_h(A + (size_t)m0 * KDIM + kk, KDIM);
    v16h bm = load_b_gen(woh + (size_t)n0 * KDIM + kk, 1, KDIM);
    acc = wmma16(a, bm, acc);
  }
  int lane = threadIdx.x & 31, n = n0 + (lane & 15), hi = lane >> 4;
#pragma unroll
  for (int rr = 0; rr < 8; ++rr)
    seqpre[(size_t)(b * NQ + m0 + rr + 8 * hi) * NHID + n] = acc[rr] + bo[n];
}

// ---------------------------------------------------------------------------
// K6: LayerNorm per (b,q) row + has-mask; K7: mean pool over Q
// ---------------------------------------------------------------------------
__global__ __launch_bounds__(256) void k_ln(
    const float* __restrict__ seqpre, const float* __restrict__ ln_g,
    const float* __restrict__ ln_b, const float* __restrict__ hasf,
    float* __restrict__ seq_out) {
  int row = blockIdx.x, b = row / NQ, t = threadIdx.x;
  __shared__ float red[256];
  float x = seqpre[(size_t)row * NHID + t];
  red[t] = x;
  __syncthreads();
  for (int s = 128; s > 0; s >>= 1) { if (t < s) red[t] += red[t + s]; __syncthreads(); }
  float mu = red[0] / (float)NHID;
  __syncthreads();
  float dv = x - mu;
  red[t] = dv * dv;
  __syncthreads();
  for (int s = 128; s > 0; s >>= 1) { if (t < s) red[t] += red[t + s]; __syncthreads(); }
  float var = red[0] / (float)NHID;
  float y = dv * rsqrtf(var + 1e-5f) * ln_g[t] + ln_b[t];
  seq_out[(size_t)row * NHID + t] = y * hasf[b];
}

__global__ __launch_bounds__(256) void k_pool(
    const float* __restrict__ seq_out, float* __restrict__ pooled) {
  int b = blockIdx.x, t = threadIdx.x;
  float s = 0.0f;
  for (int q = 0; q < NQ; ++q) s += seq_out[(size_t)(b * NQ + q) * NHID + t];
  pooled[b * NHID + t] = s / (float)NQ;
}

// ---------------------------------------------------------------------------
extern "C" void kernel_launch(void* const* d_in, const int* in_sizes, int n_in,
                              void* d_out, int out_size, void* d_ws, size_t ws_size,
                              hipStream_t stream) {
  const float* values = (const float*)d_in[0];
  const int*   mask   = (const int*)d_in[1];
  const float* times  = (const float*)d_in[2];
  const float* w_per  = (const float*)d_in[3];
  const float* b_per  = (const float*)d_in[4];
  const float* w_lin  = (const float*)d_in[5];
  const float* b_lin  = (const float*)d_in[6];
  const float* wq     = (const float*)d_in[7];
  const float* bq     = (const float*)d_in[8];
  const float* wk     = (const float*)d_in[9];
  const float* bk     = (const float*)d_in[10];
  const float* wo     = (const float*)d_in[11];
  const float* bo     = (const float*)d_in[12];
  const float* ln_g   = (const float*)d_in[13];
  const float* ln_b   = (const float*)d_in[14];

  float* out     = (float*)d_out;
  float* seq_out = out;                              // B*Q*HID
  float* pooled  = out + NB * NQ * NHID;             // B*HID
  float* qt_out  = out + NB * NQ * NHID + NB * NHID; // B*Q

  // workspace carve-out (~4.4MB; L2-resident)
  char* p = (char*)d_ws;
  auto carve = [&](size_t bytes) {
    char* r = p; p += (bytes + 255) & ~(size_t)255; return r;
  };
  float*    tn      = (float*)carve(NB * NL * 4);
  float*    hasf    = (float*)carve(NB * 4);
  float*    colmean = (float*)carve(NB * NV * 4);
  float*    seqpre  = (float*)carve((size_t)NB * NQ * NHID * 4);
  _Float16* wqh     = (_Float16*)carve(NE * NE * 2);
  _Float16* wkh     = (_Float16*)carve(NE * NE * 2);
  _Float16* woh     = (_Float16*)carve((size_t)NHID * NH * NV * 2);
  _Float16* tkh     = (_Float16*)carve((size_t)NB * NL * NE * 2);
  _Float16* tqh     = (_Float16*)carve((size_t)NQ * NE * 2);
  _Float16* qhp     = (_Float16*)carve((size_t)NH * NQ * KP * 2);
  _Float16* khp     = (_Float16*)carve((size_t)NB * NH * NL * KP * 2);
  _Float16* mcat    = (_Float16*)carve((size_t)NB * NL * NV * 2);
  _Float16* ctxh    = (_Float16*)carve((size_t)NB * NQ * NH * NV * 2);

  k_prep<<<NB, 256, 0, stream>>>(values, mask, times, tn, hasf, colmean, mcat, qt_out);
  k_cvt<<<(NHID * NH * NV + 255) / 256, 256, 0, stream>>>(wq, wk, wo, wqh, wkh, woh);
  k_embed<<<NB * NL + NQ, 128, 0, stream>>>(tn, w_per, b_per, w_lin, b_lin, tkh, tqh);
  // K projection: M = B*L = 2048 -> 1024 tiles -> 128 blocks of 8 waves
  k_proj<0><<<128, 256, 0, stream>>>(tkh, wkh, bk, khp, NB * NL, 1.0f);
  // Q projection (batch-independent): M = 128 -> 64 tiles; pre-scale 1/sqrt(EK)
  k_proj<1><<<8, 256, 0, stream>>>(tqh, wqh, bq, qhp, NQ, 0.25f);
  k_attn<<<NB * NH, 256, 0, stream>>>(qhp, khp, mcat, colmean, ctxh);
  k_ogemm<<<128, 256, 0, stream>>>(ctxh, woh, bo, seqpre);
  k_ln<<<NB * NQ, 256, 0, stream>>>(seqpre, ln_g, ln_b, hasf, seq_out);
  k_pool<<<NB, 256, 0, stream>>>(seq_out, pooled);
}
